// SIVA_82617990906061
// MI455X (gfx1250) — matrix operand order repaired and checked by reading, not compiled, not run on hardware
//
#include <hip/hip_runtime.h>
#include <hip/hip_bf16.h>

#define DEV __device__ __forceinline__

typedef __attribute__((ext_vector_type(16))) _Float16 v16h;
typedef __attribute__((ext_vector_type(8)))  _Float16 h8;
typedef __attribute__((ext_vector_type(8)))  float    v8f;

namespace cfg {
constexpr int B = 4, N = 2048, NG = 12, K = 16, H = 64, C = 128, BD = 64;
constexpr int M = N / 4;                    // 512
constexpr int R2 = B * N * NG;              // 98304 rows (embed/down gemms)
constexpr int R3 = B * M * NG * K;          // 393216 rows (kbx)
}

DEV float gelu_f(float x) {
  return 0.5f * x * (1.0f + tanhf(0.7978845608028654f * (x + 0.044715f * x * x * x)));
}

union HV16 { v16h v; h8 q[2]; _Float16 h[16]; };

// wave32 WMMA 16x16x32 f16 fragment helpers (per CDNA5 ISA 7.12.2).
// A (16x32, row = lane&15): lane halves are cols {kk+aoff+0..7, kk+aoff+16..23}
//   with aoff = (lane>=16)*8  -> two contiguous 16B runs.
// B (32x16, col = lane&15): lane halves are rows {kk+boff+0..7, kk+boff+16..23}, boff = (lane>=16)*8.
//   With W stored TRANSPOSED (WT[n][k]) that is two contiguous 16B runs.
DEV void load_a_frag(HV16& a, const _Float16* __restrict__ rowp, int kk, int lane) {
  int aoff = (lane >> 4) << 3;
  a.q[0] = *(const h8*)(rowp + kk + aoff);
  a.q[1] = *(const h8*)(rowp + kk + aoff + 16);
}
DEV void load_a_frag_lds(HV16& a, const _Float16* rowp, int kk, int lane) {
  int aoff = (lane >> 4) << 3;
  a.q[0] = *(const h8*)(rowp + kk + aoff);
  a.q[1] = *(const h8*)(rowp + kk + aoff + 16);
}
DEV void load_b_fragT(HV16& b, const _Float16* __restrict__ WT, int KdStride, int col,
                      int kk, int lane) {
  int boff = (lane >> 4) << 3;
  const _Float16* p = WT + (long)col * KdStride + kk + boff;
  b.q[0] = *(const h8*)(p);
  b.q[1] = *(const h8*)(p + 16);
}

// ---------------------------------------------------------------- utilities
__global__ void k_to_half(const float* __restrict__ in, _Float16* __restrict__ out, int n) {
  for (int i = blockIdx.x * blockDim.x + threadIdx.x; i < n; i += gridDim.x * blockDim.x)
    out[i] = (_Float16)in[i];
}

// out[n*Kd + k] = (f16) in[k*Nd + n]   (transpose for B-fragment vector loads)
__global__ void k_to_half_T(const float* __restrict__ in, _Float16* __restrict__ out,
                            int Kd, int Nd) {
  int total = Kd * Nd;
  for (int i = blockIdx.x * blockDim.x + threadIdx.x; i < total; i += gridDim.x * blockDim.x) {
    int k = i / Nd, n = i % Nd;
    out[n * Kd + k] = (_Float16)in[i];
  }
}

// ------------------------------------------------- lift + emb1 (gelu) -> f16
__global__ void k_lift_emb1(const float* __restrict__ pos, const float* __restrict__ nrm,
                            const float* __restrict__ grid, const float* __restrict__ w,
                            const float* __restrict__ bias, _Float16* __restrict__ out) {
  using namespace cfg;
  long total = (long)R2 * H;
  long stride = (long)gridDim.x * blockDim.x;
  for (long idx = blockIdx.x * (long)blockDim.x + threadIdx.x; idx < total; idx += stride) {
    int h = (int)(idx & 63);
    long row = idx >> 6;
    int ng = (int)(row % NG);
    long bn = row / NG;
    int n = (int)(bn % N);
    int b = (int)(bn / N);
    const float* p  = pos + ((long)b * N + n) * 3;
    const float* nm = nrm + ((long)b * N + n) * 3;
    const float* g  = grid + ng * 3;
    float f0 = nm[0] * g[0] + nm[1] * g[1] + nm[2] * g[2];
    float acc = bias[h] + f0 * w[h] + p[0] * w[64 + h] + p[1] * w[128 + h] + p[2] * w[192 + h];
    out[row * 64 + h] = (_Float16)gelu_f(acc);
  }
}

// ----------------------------------------- generic WMMA GEMM: act(A@W + b)
// A [R x Kd] f16 row-major, WT [Nd x Kd] f16 (transposed). blockDim = 32.
// act: 0 none, 1 relu, 2 gelu. outH!=null -> f16 out else f32 out.
__global__ void k_wmma_gemm(const _Float16* __restrict__ A, const _Float16* __restrict__ WT,
                            const float* __restrict__ bias, float* __restrict__ outF,
                            _Float16* __restrict__ outH, int Kd, int Nd, int act) {
  int lane = threadIdx.x & 31;
  int n0 = blockIdx.x * 16, r0 = blockIdx.y * 16;
  int lrow = lane & 15, ncol = lane & 15;
  const _Float16* arow = A + (long)(r0 + lrow) * Kd;
  __builtin_prefetch(arow, 0, 1);
  v8f acc = {};
  for (int kk = 0; kk < Kd; kk += 32) {
    HV16 a, bb;
    load_a_frag(a, arow, kk, lane);
    load_b_fragT(bb, WT, Kd, n0 + ncol, kk, lane);
    acc = __builtin_amdgcn_wmma_f32_16x16x32_f16(false, a.v, false, bb.v, (short)0, acc, false, false);
  }
  int c = n0 + ncol;
  float bv = bias ? bias[c] : 0.0f;
#pragma unroll
  for (int j = 0; j < 8; j++) {
    int row = r0 + j + ((lane >> 4) << 3);
    float v = acc[j] + bv;
    if (act == 1) v = fmaxf(v, 0.0f);
    else if (act == 2) v = gelu_f(v);
    if (outH) outH[(long)row * Nd + c] = (_Float16)v;
    else      outF[(long)row * Nd + c] = v;
  }
}

// ------------------------------------------------------------ FPS (serial)
__global__ void k_fps(const float* __restrict__ pos, int* __restrict__ sel) {
  using namespace cfg;
  int b = blockIdx.x;
  const float* p = pos + (long)b * N * 3;
  __shared__ float d[N];
  __shared__ float rv[256];
  __shared__ int   ri[256];
  int t = threadIdx.x;
  for (int i = t; i < N; i += 256) d[i] = 1e10f;
  __syncthreads();
  int last = 0;
  for (int m = 0; m < M; m++) {
    if (t == 0) sel[b * M + m] = last;
    float lx = p[last * 3], ly = p[last * 3 + 1], lz = p[last * 3 + 2];
    for (int i = t; i < N; i += 256) {
      float dx = p[i * 3] - lx, dy = p[i * 3 + 1] - ly, dz = p[i * 3 + 2] - lz;
      float dd = dx * dx + dy * dy + dz * dz;
      if (dd < d[i]) d[i] = dd;
    }
    __syncthreads();
    float bv = -1.0f; int bi = 0;
    for (int i = t; i < N; i += 256) {
      float v = d[i];
      if (v > bv || (v == bv && i < bi)) { bv = v; bi = i; }
    }
    rv[t] = bv; ri[t] = bi;
    __syncthreads();
    for (int s = 128; s > 0; s >>= 1) {
      if (t < s && (rv[t + s] > rv[t] || (rv[t + s] == rv[t] && ri[t + s] < ri[t]))) {
        rv[t] = rv[t + s]; ri[t] = ri[t + s];
      }
      __syncthreads();
    }
    last = ri[0];
    __syncthreads();
  }
}

// ------------------------------------------------------ KNN (block/query)
__global__ void k_knn(const float* __restrict__ pos, const int* __restrict__ selq,
                      const int* __restrict__ selp, int Q, int P, int* __restrict__ out) {
  using namespace cfg;
  int b = blockIdx.x / Q, q = blockIdx.x % Q;
  const float* pb = pos + (long)b * N * 3;
  __shared__ float dist[N];
  __shared__ float rv[256];
  __shared__ int   ri[256];
  int t = threadIdx.x;
  int qi = selq ? selq[b * M + q] : q;
  float qx = pb[qi * 3], qy = pb[qi * 3 + 1], qz = pb[qi * 3 + 2];
  for (int i = t; i < P; i += 256) {
    int pi = selp ? selp[b * M + i] : i;
    float dx = pb[pi * 3] - qx, dy = pb[pi * 3 + 1] - qy, dz = pb[pi * 3 + 2] - qz;
    dist[i] = dx * dx + dy * dy + dz * dz;
  }
  __syncthreads();
  for (int k = 0; k < K; k++) {
    float bv = 3.0e38f; int bi = 0;
    for (int i = t; i < P; i += 256) {
      float v = dist[i];
      if (v < bv || (v == bv && i < bi)) { bv = v; bi = i; }
    }
    rv[t] = bv; ri[t] = bi;
    __syncthreads();
    for (int s = 128; s > 0; s >>= 1) {
      if (t < s && (rv[t + s] < rv[t] || (rv[t + s] == rv[t] && ri[t + s] < ri[t]))) {
        rv[t] = rv[t + s]; ri[t] = ri[t + s];
      }
      __syncthreads();
    }
    if (t == 0) {
      out[((long)b * Q + q) * K + k] = ri[0];
      dist[ri[0]] = 3.4e38f;
    }
    __syncthreads();
  }
}

// ----------------------------------------------------------- scatter max
__global__ void k_scatter_max(const float* __restrict__ Hbuf, const int* __restrict__ nbrd,
                              float* __restrict__ xout) {
  using namespace cfg;
  long total = (long)B * M * NG * C;
  long stride = (long)gridDim.x * blockDim.x;
  for (long idx = blockIdx.x * (long)blockDim.x + threadIdx.x; idx < total; idx += stride) {
    int c = (int)(idx & 127);
    long r = idx >> 7;
    int ng = (int)(r % NG);
    long bm = r / NG;
    int m = (int)(bm % M);
    int b = (int)(bm / M);
    const int* nb = nbrd + ((long)b * M + m) * K;
    float v = -3.4e38f;
    for (int k = 0; k < K; k++) {
      int src = nb[k];
      float h = Hbuf[(((long)b * N + src) * NG + ng) * C + c];
      v = fmaxf(v, h);
    }
    xout[idx] = v;
  }
}

// ------------------------------------------ edge attrs rpx [B][M][NG][K][2]
__global__ void k_edges(const float* __restrict__ pos, const int* __restrict__ sel,
                        const int* __restrict__ nbr, const float* __restrict__ grid,
                        float* __restrict__ rpx) {
  using namespace cfg;
  long total = (long)B * M * NG * K;
  long stride = (long)gridDim.x * blockDim.x;
  for (long idx = blockIdx.x * (long)blockDim.x + threadIdx.x; idx < total; idx += stride) {
    int k = (int)(idx & 15);
    long r = idx >> 4;
    int ng = (int)(r % NG);
    long bm = r / NG;
    int m = (int)(bm % M);
    int b = (int)(bm / M);
    const float* pb = pos + (long)b * N * 3;
    int mi = sel[b * M + m];
    int si = sel[b * M + nbr[((long)b * M + m) * K + k]];
    float rx = pb[si * 3]     - pb[mi * 3];
    float ry = pb[si * 3 + 1] - pb[mi * 3 + 1];
    float rz3 = pb[si * 3 + 2] - pb[mi * 3 + 2];
    const float* g = grid + ng * 3;
    float rz = rx * g[0] + ry * g[1] + rz3 * g[2];
    float ex = rx - rz * g[0], ey = ry - rz * g[1], ez = rz3 - rz * g[2];
    float rxy = sqrtf(ex * ex + ey * ey + ez * ez + 1e-12f);
    rpx[idx * 2 + 0] = rxy;
    rpx[idx * 2 + 1] = rz;
  }
}

// ----------------------------------- orientation basis: RFF features [B*144][128]
__global__ void k_fR(const float* __restrict__ grid, const float* __restrict__ rff,
                     float* __restrict__ fR) {
  using namespace cfg;
  long total = (long)B * NG * NG * 128;
  long stride = (long)gridDim.x * blockDim.x;
  for (long idx = blockIdx.x * (long)blockDim.x + threadIdx.x; idx < total; idx += stride) {
    int j = (int)(idx & 127);
    long r = idx >> 7;
    int rg = (int)(r % (NG * NG));
    int a = rg / NG, d = rg % NG;
    float z = grid[a * 3] * grid[d * 3] + grid[a * 3 + 1] * grid[d * 3 + 1] +
              grid[a * 3 + 2] * grid[d * 3 + 2];
    int jj = j & 63;
    float p = 6.283185307179586f * z * rff[jj];
    fR[idx] = (j < 64) ? sinf(p) : cosf(p);
  }
}

// --------------------------------------------- small VALU GEMM act(A@W + b)
__global__ void k_small_mm(const float* __restrict__ A, const float* __restrict__ W,
                           const float* __restrict__ bias, float* __restrict__ out,
                           int rows, int Kd, int Nd, int act) {
  long total = (long)rows * Nd;
  long stride = (long)gridDim.x * blockDim.x;
  for (long idx = blockIdx.x * (long)blockDim.x + threadIdx.x; idx < total; idx += stride) {
    int c = (int)(idx % Nd);
    long r = idx / Nd;
    float acc = bias[c];
    for (int kk = 0; kk < Kd; kk++) acc += A[r * Kd + kk] * W[(long)kk * Nd + c];
    if (act == 2) acc = gelu_f(acc);
    out[idx] = acc;
  }
}

// ------------- fused kbx: RFF(sin/cos) -> WMMA bx1 -> gelu -> WMMA bx2 -> gelu
// 4 waves/block, each wave owns one 16-row tile (= one (b,m,ng), rows = k).
// w1t [128(n) x 128(k)], w2t [64(n) x 128(k)]  (transposed f16)
__global__ void k_kbx(const float* __restrict__ rpx, const float* __restrict__ rff,
                      const _Float16* __restrict__ w1t, const float* __restrict__ b1,
                      const _Float16* __restrict__ w2t, const float* __restrict__ b2,
                      _Float16* __restrict__ kbx) {
  using namespace cfg;
  __shared__ __align__(16) _Float16 hbuf[4][16][128];
  int wave = threadIdx.x >> 5, lane = threadIdx.x & 31;
  int tile = blockIdx.x * 4 + wave;          // < B*M*NG
  long row0 = (long)tile * 16;
  int lrow = lane & 15, ncol = lane & 15;
  float z0 = rpx[(row0 + lrow) * 2 + 0];
  float z1 = rpx[(row0 + lrow) * 2 + 1];
  // bx1: [16x128] = sincos(rff) @ w1  (Kd=128, Nd=128)
  for (int nt = 0; nt < 8; nt++) {
    v8f acc = {};
    for (int kk = 0; kk < 128; kk += 32) {
      HV16 a, bb;
      int aoff = (lane >> 4) << 3;
#pragma unroll
      for (int e = 0; e < 16; e++) {
        int col = kk + (e & 7) + ((e >> 3) << 4) + aoff;
        int j = col & 63;
        float p = 6.283185307179586f * (z0 * rff[j] + z1 * rff[64 + j]);
        a.h[e] = (_Float16)((col < 64) ? sinf(p) : cosf(p));
      }
      load_b_fragT(bb, w1t, 128, nt * 16 + ncol, kk, lane);
      acc = __builtin_amdgcn_wmma_f32_16x16x32_f16(false, a.v, false, bb.v, (short)0, acc, false, false);
    }
    int c = nt * 16 + ncol;
    float bv = b1[c];
#pragma unroll
    for (int j = 0; j < 8; j++) {
      int r = j + ((lane >> 4) << 3);
      hbuf[wave][r][c] = (_Float16)gelu_f(acc[j] + bv);
    }
  }
  __syncthreads();
  // bx2: [16x64] = h @ w2 (Kd=128, Nd=64)
  for (int nt = 0; nt < 4; nt++) {
    v8f acc = {};
    for (int kk = 0; kk < 128; kk += 32) {
      HV16 a, bb;
      load_a_frag_lds(a, &hbuf[wave][lrow][0], kk, lane);
      load_b_fragT(bb, w2t, 128, nt * 16 + ncol, kk, lane);
      acc = __builtin_amdgcn_wmma_f32_16x16x32_f16(false, a.v, false, bb.v, (short)0, acc, false, false);
    }
    int c = nt * 16 + ncol;
    float bv = b2[c];
#pragma unroll
    for (int j = 0; j < 8; j++) {
      long r = row0 + j + ((lane >> 4) << 3);
      kbx[r * 64 + c] = (_Float16)gelu_f(acc[j] + bv);
    }
  }
}

// --------------------- interaction layer: one block per (b,m), 4 waves/block
// kxwt [128(n) x 64(k)], mixwt [128(n) x 128(k)]  (transposed f16)
__global__ void k_interaction(const float* __restrict__ xin, float* __restrict__ xout,
                              const _Float16* __restrict__ kbx, const int* __restrict__ nbr,
                              const float* __restrict__ kR, const _Float16* __restrict__ kxwt,
                              const float* __restrict__ kxb, const _Float16* __restrict__ mixwt,
                              const float* __restrict__ mixb) {
  using namespace cfg;
  __shared__ int   nb[16];
  __shared__ float agg[NG][C];
  __shared__ __align__(16) _Float16 ybufh[16][C];
  int b = blockIdx.x / M, m = blockIdx.x % M;
  int t = threadIdx.x, wave = t >> 5, lane = t & 31;
  int lrow = lane & 15, ncol = lane & 15;
  if (t < 16) nb[t] = nbr[((long)b * M + m) * K + t];
  // zero the pad rows 12..15 of y
  for (int i = t; i < 4 * C; i += 128) ybufh[12 + (i >> 7)][i & 127] = (_Float16)0.0f;
  __syncthreads();

  // agg[ng][c] = mean_k (kbx[m,ng,k]@kx_w + b)[c] * x[nbr[k]][ng][c]
  for (int job = wave; job < NG * 8; job += 4) {
    int ng = job >> 3, nt = job & 7;
    long rbase = (((long)b * M + m) * NG + ng) * K;
    const _Float16* arow = kbx + (rbase + lrow) * 64;
    __builtin_prefetch(arow, 0, 1);
    v8f acc = {};
    for (int kk = 0; kk < 64; kk += 32) {
      HV16 a, bb;
      load_a_frag(a, arow, kk, lane);
      load_b_fragT(bb, kxwt, 64, nt * 16 + ncol, kk, lane);
      acc = __builtin_amdgcn_wmma_f32_16x16x32_f16(false, a.v, false, bb.v, (short)0, acc, false, false);
    }
    int c = nt * 16 + ncol;
    float bv = kxb[c];
    float part = 0.0f;
#pragma unroll
    for (int j = 0; j < 8; j++) {
      int k = j + ((lane >> 4) << 3);
      float xv = xin[(((long)b * M + nb[k]) * NG + ng) * C + c];
      part += (acc[j] + bv) * xv;
    }
    part += __shfl_xor(part, 16, 32);
    if (lane < 16) agg[ng][c] = part * (1.0f / (float)K);
  }
  __syncthreads();

  // orientation mix: y[ng][c] = (1/NG) sum_n agg[n][c] * kR[ng][n][c]
  const float* kRg = kR + (long)b * NG * NG * C;
  for (int i = t; i < NG * C; i += 128) {
    int ng = i >> 7, c = i & 127;
    float s = 0.0f;
    for (int n = 0; n < NG; n++) s += agg[n][c] * kRg[((long)ng * NG + n) * C + c];
    ybufh[ng][c] = (_Float16)(s * (1.0f / (float)NG));
  }
  __syncthreads();

  // mix GEMM (rows 12..15 zero padded) + residual
  for (int nt = wave * 2; nt < wave * 2 + 2; nt++) {
    v8f acc = {};
    for (int kk = 0; kk < 128; kk += 32) {
      HV16 a, bb;
      load_a_frag_lds(a, &ybufh[lrow][0], kk, lane);
      load_b_fragT(bb, mixwt, 128, nt * 16 + ncol, kk, lane);
      acc = __builtin_amdgcn_wmma_f32_16x16x32_f16(false, a.v, false, bb.v, (short)0, acc, false, false);
    }
    int c = nt * 16 + ncol;
    float bv = mixb[c];
#pragma unroll
    for (int j = 0; j < 8; j++) {
      int ng = j + ((lane >> 4) << 3);
      if (ng < NG) {
        long idx = (((long)b * M + m) * NG + ng) * C + c;
        xout[idx] = xin[idx] + gelu_f(acc[j] + bv);
      }
    }
  }
}

// ------------------------------------------------ readout (one block/graph)
__global__ void k_readout(const float* __restrict__ xf, const float* __restrict__ p1w,
                          const float* __restrict__ p1b, const float* __restrict__ p2w,
                          const float* __restrict__ p2b, const float* __restrict__ p3w,
                          const float* __restrict__ p3b, float* __restrict__ out) {
  using namespace cfg;
  int b = blockIdx.x, t = threadIdx.x;     // 256 threads
  __shared__ float g[C];
  __shared__ float h1[256];
  __shared__ float h2[64];
  if (t < C) {
    float s = 0.0f;
    const float* base = xf + (long)b * M * NG * C + t;
    for (int r = 0; r < M * NG; r++) s += base[(long)r * C];
    g[t] = s / (float)(M * NG);
  }
  __syncthreads();
  {
    float s = p1b[t];
    for (int c = 0; c < C; c++) s += g[c] * p1w[c * 256 + t];
    h1[t] = gelu_f(s);
  }
  __syncthreads();
  if (t < 64) {
    float s = p2b[t];
    for (int c = 0; c < 256; c++) s += h1[c] * p2w[c * 64 + t];
    h2[t] = gelu_f(s);
  }
  __syncthreads();
  if (t == 0) {
    float s = p3b[0];
    for (int c = 0; c < 64; c++) s += h2[c] * p3w[c];
    out[b] = s;
  }
}

// ===========================================================================
extern "C" void kernel_launch(void* const* d_in, const int* in_sizes, int n_in,
                              void* d_out, int out_size, void* d_ws, size_t ws_size,
                              hipStream_t stream) {
  using namespace cfg;
  (void)in_sizes; (void)n_in; (void)out_size; (void)ws_size;
  const float* pos     = (const float*)d_in[0];
  const float* normal  = (const float*)d_in[1];
  const float* grid    = (const float*)d_in[2];
  const float* emb1_w  = (const float*)d_in[3];
  const float* emb1_b  = (const float*)d_in[4];
  const float* emb2_w  = (const float*)d_in[5];
  const float* emb2_b  = (const float*)d_in[6];
  const float* down_w  = (const float*)d_in[7];
  const float* down_b  = (const float*)d_in[8];
  const float* bx_rff  = (const float*)d_in[9];
  const float* bx1_w   = (const float*)d_in[10];
  const float* bx1_b   = (const float*)d_in[11];
  const float* bx2_w   = (const float*)d_in[12];
  const float* bx2_b   = (const float*)d_in[13];
  const float* bR_rff  = (const float*)d_in[14];
  const float* bR1_w   = (const float*)d_in[15];
  const float* bR1_b   = (const float*)d_in[16];
  const float* bR2_w   = (const float*)d_in[17];
  const float* bR2_b   = (const float*)d_in[18];
  const float* i1_kx_w = (const float*)d_in[19];
  const float* i1_kx_b = (const float*)d_in[20];
  const float* i1_kR_w = (const float*)d_in[21];
  const float* i1_kR_b = (const float*)d_in[22];
  const float* i1_mx_w = (const float*)d_in[23];
  const float* i1_mx_b = (const float*)d_in[24];
  const float* i2_kx_w = (const float*)d_in[25];
  const float* i2_kx_b = (const float*)d_in[26];
  const float* i2_kR_w = (const float*)d_in[27];
  const float* i2_kR_b = (const float*)d_in[28];
  const float* i2_mx_w = (const float*)d_in[29];
  const float* i2_mx_b = (const float*)d_in[30];
  const float* p1w = (const float*)d_in[31];
  const float* p1b = (const float*)d_in[32];
  const float* p2w = (const float*)d_in[33];
  const float* p2b = (const float*)d_in[34];
  const float* p3w = (const float*)d_in[35];
  const float* p3b = (const float*)d_in[36];

  char* ws = (char*)d_ws;
  size_t off = 0;
  auto take = [&](size_t bytes) -> void* {
    void* p = ws + off;
    off += (bytes + 255) & ~(size_t)255;
    return p;
  };
  _Float16* A1h  = (_Float16*)take((size_t)R2 * 64 * 2);
  _Float16* X0h  = (_Float16*)take((size_t)R2 * 64 * 2);
  float*    Hf   = (float*)take((size_t)R2 * 128 * 4);
  int*      sel  = (int*)take((size_t)B * M * 4);
  int*      nbrd = (int*)take((size_t)B * M * K * 4);
  int*      nbr  = (int*)take((size_t)B * M * K * 4);
  float*    xA   = (float*)take((size_t)B * M * NG * C * 4);
  float*    xB   = (float*)take((size_t)B * M * NG * C * 4);
  float*    rpx  = (float*)take((size_t)B * M * NG * K * 2 * 4);
  _Float16* kbxh = (_Float16*)take((size_t)R3 * 64 * 2);
  float*    fR   = (float*)take((size_t)B * 144 * 128 * 4);
  float*    hR   = (float*)take((size_t)B * 144 * 128 * 4);
  float*    kbR  = (float*)take((size_t)B * 144 * 64 * 4);
  float*    kR1  = (float*)take((size_t)B * 144 * 128 * 4);
  float*    kR2  = (float*)take((size_t)B * 144 * 128 * 4);
  _Float16* emb2t = (_Float16*)take(64 * 64 * 2);
  _Float16* downt = (_Float16*)take(64 * 128 * 2);
  _Float16* bx1t  = (_Float16*)take(128 * 128 * 2);
  _Float16* bx2t  = (_Float16*)take(128 * 64 * 2);
  _Float16* kx1t  = (_Float16*)take(64 * 128 * 2);
  _Float16* mx1t  = (_Float16*)take(128 * 128 * 2);
  _Float16* kx2t  = (_Float16*)take(64 * 128 * 2);
  _Float16* mx2t  = (_Float16*)take(128 * 128 * 2);

  auto cvtT = [&](const float* src, _Float16* dst, int Kd, int Nd) {
    int n = Kd * Nd;
    k_to_half_T<<<(n + 255) / 256, 256, 0, stream>>>(src, dst, Kd, Nd);
  };
  cvtT(emb2_w, emb2t, 64, 64);     // -> [64 n][64 k]
  cvtT(down_w, downt, 64, 128);    // -> [128][64]
  cvtT(bx1_w,  bx1t, 128, 128);    // -> [128][128]
  cvtT(bx2_w,  bx2t, 128, 64);     // -> [64][128]
  cvtT(i1_kx_w, kx1t, 64, 128);    // -> [128][64]
  cvtT(i1_mx_w, mx1t, 128, 128);   // -> [128][128]
  cvtT(i2_kx_w, kx2t, 64, 128);    // -> [128][64]
  cvtT(i2_mx_w, mx2t, 128, 128);   // -> [128][128]

  // lift + emb1
  k_lift_emb1<<<4096, 256, 0, stream>>>(pos, normal, grid, emb1_w, emb1_b, A1h);
  // emb2 (WMMA, no act, f16 out)
  k_wmma_gemm<<<dim3(64 / 16, R2 / 16), 32, 0, stream>>>(A1h, emb2t, emb2_b, nullptr, X0h, 64, 64, 0);
  // down (WMMA, relu, f32 out)
  k_wmma_gemm<<<dim3(128 / 16, R2 / 16), 32, 0, stream>>>(X0h, downt, down_b, Hf, nullptr, 64, 128, 1);
  // FPS + KNNs
  k_fps<<<B, 256, 0, stream>>>(pos, sel);
  k_knn<<<B * M, 256, 0, stream>>>(pos, sel, nullptr, M, N, nbrd);
  k_scatter_max<<<4096, 256, 0, stream>>>(Hf, nbrd, xA);
  k_knn<<<B * M, 256, 0, stream>>>(pos, sel, sel, M, M, nbr);
  // edge attrs
  k_edges<<<1536, 256, 0, stream>>>(pos, sel, nbr, grid, rpx);
  // orientation basis + per-layer kR (small, VALU)
  k_fR<<<288, 256, 0, stream>>>(grid, bR_rff, fR);
  k_small_mm<<<288, 256, 0, stream>>>(fR, bR1_w, bR1_b, hR, B * 144, 128, 128, 2);
  k_small_mm<<<144, 256, 0, stream>>>(hR, bR2_w, bR2_b, kbR, B * 144, 128, 64, 2);
  k_small_mm<<<288, 256, 0, stream>>>(kbR, i1_kR_w, i1_kR_b, kR1, B * 144, 64, 128, 0);
  k_small_mm<<<288, 256, 0, stream>>>(kbR, i2_kR_w, i2_kR_b, kR2, B * 144, 64, 128, 0);
  // edge basis MLP (fused RFF + 2x WMMA)
  k_kbx<<<(B * M * NG) / 4, 128, 0, stream>>>(rpx, bx_rff, bx1t, bx1_b, bx2t, bx2_b, kbxh);
  // interaction layers
  k_interaction<<<B * M, 128, 0, stream>>>(xA, xB, kbxh, nbr, kR1, kx1t, i1_kx_b, mx1t, i1_mx_b);
  k_interaction<<<B * M, 128, 0, stream>>>(xB, xA, kbxh, nbr, kR2, kx2t, i2_kx_b, mx2t, i2_mx_b);
  // readout
  k_readout<<<B, 256, 0, stream>>>(xA, p1w, p1b, p2w, p2b, p3w, p3b, (float*)d_out);
}
